// TwoLayerPerceptron_norm2_77180562309360
// MI455X (gfx1250) — matrix operand order repaired and checked by reading, not compiled, not run on hardware
//
#include <hip/hip_runtime.h>
#include <hip/hip_bf16.h>

typedef __attribute__((ext_vector_type(16))) __bf16 v16bf;
typedef __attribute__((ext_vector_type(8)))  __bf16 v8bf;
typedef __attribute__((ext_vector_type(8)))  float  v8f;

#define BB   16384
#define INF_ 1024
#define D1_  8
#define H1_  2048
#define H2_  1024
#define NC_  1000
#define NCP_ 1024   // padded
#define C_   128
#define EPS_ 1e-5f

// ---------------------------------------------------------------------------
// Async global->LDS staging (CDNA5 ASYNCcnt path) with portable fallback.
// Builtin signature (from hipcc diagnostic): v4i AS1* src, v4i AS3* dst, Ii, Ii.
// ---------------------------------------------------------------------------
#if __has_builtin(__builtin_amdgcn_global_load_async_to_lds_b128)
#define GEMM_ASYNC 1
typedef int v4i_ __attribute__((vector_size(16)));
typedef __attribute__((address_space(1))) v4i_ as1_v4i;
typedef __attribute__((address_space(3))) v4i_ as3_v4i;
#else
#define GEMM_ASYNC 0
#endif

__device__ __forceinline__ void cp16_to_lds(__bf16* ldst, const __bf16* gsrc) {
#if GEMM_ASYNC
    // generic LDS pointers carry the LDS byte offset in their low 32 bits
    __builtin_amdgcn_global_load_async_to_lds_b128(
        (as1_v4i*)(uintptr_t)gsrc,
        (as3_v4i*)(unsigned)(uintptr_t)ldst,
        0, 0);
#else
    *(v8bf*)ldst = *(const v8bf*)gsrc;
#endif
}

__device__ __forceinline__ void wait_async_done() {
#if GEMM_ASYNC
#if __has_builtin(__builtin_amdgcn_s_wait_asynccnt)
    __builtin_amdgcn_s_wait_asynccnt(0);
#else
    asm volatile("s_wait_asynccnt 0" ::: "memory");
#endif
#endif
}

// ---------------------------------------------------------------------------
// Column statistics (two-pass, deterministic): partial sums over row chunks.
// ---------------------------------------------------------------------------
__global__ void colstats_partial(const float* __restrict__ Y, int ncols,
                                 int rows_per_chunk,
                                 float* __restrict__ psum, float* __restrict__ psq) {
    int col = blockIdx.x * blockDim.x + threadIdx.x;
    int r0  = blockIdx.y * rows_per_chunk;
    float s = 0.f, q = 0.f;
    for (int r = r0; r < r0 + rows_per_chunk; ++r) {
        float v = Y[(size_t)r * ncols + col];
        s += v; q += v * v;
    }
    psum[(size_t)blockIdx.y * ncols + col] = s;
    psq [(size_t)blockIdx.y * ncols + col] = q;
}

__global__ void colstats_final(const float* __restrict__ psum, const float* __restrict__ psq,
                               int nchunks, int ncols, float invM,
                               const float* __restrict__ g, const float* __restrict__ b,
                               float* __restrict__ scale, float* __restrict__ shift) {
    int col = blockIdx.x * blockDim.x + threadIdx.x;
    float s = 0.f, q = 0.f;
    for (int c = 0; c < nchunks; ++c) {
        s += psum[(size_t)c * ncols + col];
        q += psq [(size_t)c * ncols + col];
    }
    float mean = s * invM;
    float var  = q * invM - mean * mean;
    float rs   = rsqrtf(var + EPS_);
    float sc   = g[col] * rs;
    scale[col] = sc;
    shift[col] = b[col] - mean * sc;
}

// ---------------------------------------------------------------------------
// Fused BN0 + per-channel MM block (STE forward): one thread per (b,c).
// ---------------------------------------------------------------------------
__global__ void mm_block(const float* __restrict__ x,
                         const float* __restrict__ sc0, const float* __restrict__ sh0,
                         const float* __restrict__ S,  const float* __restrict__ Hm,
                         const float* __restrict__ T,  const float* __restrict__ LUT,
                         __bf16* __restrict__ vb) {
    int gid = blockIdx.x * blockDim.x + threadIdx.x;   // b*128 + c
    int c = gid & (C_ - 1);
    int b = gid >> 7;

    const float* xr = x + (size_t)b * INF_ + c * D1_;
    float h[D1_];
    #pragma unroll
    for (int d = 0; d < D1_; ++d)
        h[d] = xr[d] * sc0[c * D1_ + d] + sh0[c * D1_ + d];

    float q[15];
    const float* Sc = S + (size_t)c * D1_ * 15;
    const float* Tc = T + (size_t)c * 15;
    #pragma unroll
    for (int k = 0; k < 15; ++k) {
        float z = -Tc[k] - 1e-4f;
        #pragma unroll
        for (int d = 0; d < D1_; ++d) z += h[d] * Sc[d * 15 + k];
        q[k] = (z > 0.f) ? 1.f : ((z < 0.f) ? -1.f : 0.f);
    }

    float best = -1e30f; int jmax = 0;
    #pragma unroll
    for (int j = 0; j < 16; ++j) {
        float l = 0.f;
        #pragma unroll
        for (int k = 0; k < 15; ++k) l += q[k] * Hm[k * 16 + j];
        if (l > best) { best = l; jmax = j; }
    }

    const float* lr = LUT + (size_t)c * 16 * D1_ + jmax * D1_;
    v8bf o;
    #pragma unroll
    for (int d = 0; d < D1_; ++d) o[d] = (__bf16)lr[d];
    *(v8bf*)(vb + (size_t)b * INF_ + c * D1_) = o;
}

// ---------------------------------------------------------------------------
// f32 -> bf16 weight conversion with optional zero row-padding.
// ---------------------------------------------------------------------------
__global__ void convert_pad(const float* __restrict__ src, __bf16* __restrict__ dst,
                            int src_rows, int cols, long total) {
    long idx = (long)blockIdx.x * blockDim.x + threadIdx.x;
    if (idx >= total) return;
    long row = idx / cols, col = idx - row * cols;
    float v = (row < src_rows) ? src[row * (long)cols + col] : 0.f;
    dst[idx] = (__bf16)v;
}

// ---------------------------------------------------------------------------
// Tiled WMMA GEMM:  C[M,N] = A[M,K](bf16) * W[N,K]^T(bf16) + bias  (f32 out)
// 256 threads = 8 waves; block tile 128x128; wave tile 32x64; BK=32.
// Double-buffered LDS fed by async global->LDS (ASYNCcnt) when available.
// ---------------------------------------------------------------------------
#define LDSS 48   // padded K-stride (bf16 elems); 96B rows keep 16B alignment

__global__ __launch_bounds__(256)
void wmma_gemm_bf16(const __bf16* __restrict__ A, const __bf16* __restrict__ Wt,
                    float* __restrict__ Cout, int K, int N,
                    const float* __restrict__ bias, int nbias) {
    __shared__ __align__(16) __bf16 As[2][128 * LDSS];
    __shared__ __align__(16) __bf16 Bs[2][128 * LDSS];

    const int tid  = threadIdx.x;
    const int lane = tid & 31;
    const int wave = tid >> 5;
    const int wm   = (wave & 3) * 32;   // wave M offset in block tile
    const int wn   = (wave >> 2) * 64;  // wave N offset in block tile
    const size_t blockM = (size_t)blockIdx.y * 128;
    const size_t blockN = (size_t)blockIdx.x * 128;

    const int ldrow = tid >> 1;         // 0..127: tile row loaded by this thread
    const int khalf = (tid & 1) * 16;   // 16-element half of the 32-wide K slab
    const int ma    = lane & 15;
    const int ksel  = lane >> 4;

    const __bf16* agBase = A  + (blockM + ldrow) * (size_t)K + khalf;
    const __bf16* bgBase = Wt + (blockN + ldrow) * (size_t)K + khalf;
    const int ldsOff = ldrow * LDSS + khalf;

    auto stage = [&](int buf, int k0) {
        cp16_to_lds(&As[buf][ldsOff],     agBase + k0);
        cp16_to_lds(&As[buf][ldsOff + 8], agBase + k0 + 8);
        cp16_to_lds(&Bs[buf][ldsOff],     bgBase + k0);
        cp16_to_lds(&Bs[buf][ldsOff + 8], bgBase + k0 + 8);
    };

    v8f acc[2][4];
    #pragma unroll
    for (int i = 0; i < 2; ++i)
        #pragma unroll
        for (int j = 0; j < 4; ++j)
            #pragma unroll
            for (int r = 0; r < 8; ++r) acc[i][j][r] = 0.f;

    const int nsteps = K >> 5;
    stage(0, 0);   // prologue

    for (int s = 0; s < nsteps; ++s) {
        wait_async_done();       // own async tile writes complete
        __syncthreads();         // all waves' writes published / prior reads done
        const int cur = s & 1;
        if (s + 1 < nsteps) stage(cur ^ 1, (s + 1) << 5);

        // ---- build fragments (ISA 7.12.2 16-bit layouts) ----
        v16bf afr[2], bfr[4];
        #pragma unroll
        for (int i = 0; i < 2; ++i) {
            int m = wm + i * 16 + ma;
            ((v8bf*)&afr[i])[0] = *(const v8bf*)&As[cur][m * LDSS + ksel * 8];
            ((v8bf*)&afr[i])[1] = *(const v8bf*)&As[cur][m * LDSS + 16 + ksel * 8];
        }
        #pragma unroll
        for (int j = 0; j < 4; ++j) {
            int n = wn + j * 16 + ma;
            ((v8bf*)&bfr[j])[0] = *(const v8bf*)&Bs[cur][n * LDSS + ksel * 8];
            ((v8bf*)&bfr[j])[1] = *(const v8bf*)&Bs[cur][n * LDSS + 16 + ksel * 8];
        }

        // ---- 8 WMMAs per wave per K-step ----
        #pragma unroll
        for (int i = 0; i < 2; ++i)
            #pragma unroll
            for (int j = 0; j < 4; ++j)
                acc[i][j] = __builtin_amdgcn_wmma_f32_16x16x32_bf16(
                    false, afr[i], false, bfr[j], (short)0, acc[i][j], false, false);
    }

    // ---- epilogue: C-frag layout (VGPR r -> M=r / r+8; N = lane&15) ----
    float bv[4];
    #pragma unroll
    for (int j = 0; j < 4; ++j) {
        int col = (int)blockN + wn + j * 16 + ma;
        bv[j] = (col < nbias) ? bias[col] : 0.f;
    }
    #pragma unroll
    for (int i = 0; i < 2; ++i) {
        const size_t mbase = blockM + wm + i * 16 + ksel * 8;
        float* cp = Cout + mbase * (size_t)N + (blockN + wn + ma);
        #pragma unroll
        for (int j = 0; j < 4; ++j)
            #pragma unroll
            for (int r = 0; r < 8; ++r)
                cp[(size_t)r * N + j * 16] = acc[i][j][r] + bv[j];
    }
}

// ---------------------------------------------------------------------------
// BN + ReLU epilogue; writes f32 (optional) and bf16 (next-GEMM input).
// ---------------------------------------------------------------------------
__global__ void bnrelu(const float* __restrict__ Y,
                       const float* __restrict__ scale, const float* __restrict__ shift,
                       int ncols, long total,
                       float* __restrict__ out_f32, __bf16* __restrict__ out_bf) {
    long idx = (long)blockIdx.x * blockDim.x + threadIdx.x;
    if (idx >= total) return;
    int col = (int)(idx % ncols);
    float v = fmaxf(Y[idx] * scale[col] + shift[col], 0.f);
    if (out_f32) out_f32[idx] = v;
    out_bf[idx] = (__bf16)v;
}

// ---------------------------------------------------------------------------
// Row-wise log-softmax over the first NC_ columns (ld = padded width).
// ---------------------------------------------------------------------------
__global__ void logsoftmax_rows(const float* __restrict__ Y, int ld,
                                float* __restrict__ out) {
    __shared__ float sred[256];
    const int tid = threadIdx.x;
    const float* row = Y + (size_t)blockIdx.x * ld;

    float m = -1e30f;
    for (int j = tid; j < NC_; j += 256) m = fmaxf(m, row[j]);
    sred[tid] = m; __syncthreads();
    for (int s = 128; s > 0; s >>= 1) {
        if (tid < s) sred[tid] = fmaxf(sred[tid], sred[tid + s]);
        __syncthreads();
    }
    float rmax = sred[0]; __syncthreads();

    float ssum = 0.f;
    for (int j = tid; j < NC_; j += 256) ssum += expf(row[j] - rmax);
    sred[tid] = ssum; __syncthreads();
    for (int s = 128; s > 0; s >>= 1) {
        if (tid < s) sred[tid] += sred[tid + s];
        __syncthreads();
    }
    float lse = logf(sred[0]);

    float* orow = out + (size_t)blockIdx.x * NC_;
    for (int j = tid; j < NC_; j += 256) orow[j] = row[j] - rmax - lse;
}

// ---------------------------------------------------------------------------
extern "C" void kernel_launch(void* const* d_in, const int* in_sizes, int n_in,
                              void* d_out, int out_size, void* d_ws, size_t ws_size,
                              hipStream_t stream) {
    (void)in_sizes; (void)n_in; (void)out_size; (void)ws_size;
    const float* x       = (const float*)d_in[0];
    const float* norm0_g = (const float*)d_in[1];
    const float* norm0_b = (const float*)d_in[2];
    const float* S       = (const float*)d_in[3];
    const float* Hm      = (const float*)d_in[4];
    const float* T       = (const float*)d_in[5];
    const float* LUT     = (const float*)d_in[6];
    const float* W1      = (const float*)d_in[7];
    const float* b1      = (const float*)d_in[8];
    const float* norm1_g = (const float*)d_in[9];
    const float* norm1_b = (const float*)d_in[10];
    const float* W2      = (const float*)d_in[11];
    const float* b2      = (const float*)d_in[12];
    const float* norm2_g = (const float*)d_in[13];
    const float* norm2_b = (const float*)d_in[14];
    const float* W3      = (const float*)d_in[15];
    const float* b3      = (const float*)d_in[16];

    // --- workspace layout (256B aligned), with buffer reuse ---
    char* ws = (char*)d_ws;
    size_t off = 0;
    auto alloc = [&](size_t bytes) { size_t o = off; off = (off + bytes + 255) & ~(size_t)255; return o; };
    __bf16* bufA = (__bf16*)(ws + alloc((size_t)BB * INF_ * 2));   // v bf16 / h2 bf16
    float*  bufB = (float* )(ws + alloc((size_t)BB * H1_  * 4));   // y1 / y2
    char*   bufC =           ws + alloc((size_t)BB * H1_  * 2);    // h1 bf16 / y3 f32
    __bf16* w1b  = (__bf16*)(ws + alloc((size_t)H1_ * INF_ * 2));
    __bf16* w2b  = (__bf16*)(ws + alloc((size_t)H2_ * H1_  * 2));
    __bf16* w3b  = (__bf16*)(ws + alloc((size_t)NCP_ * H2_ * 2));
    float*  psum = (float* )(ws + alloc((size_t)64 * H1_ * 4));
    float*  psq  = (float* )(ws + alloc((size_t)64 * H1_ * 4));
    float*  sc0  = (float* )(ws + alloc(INF_ * 4));
    float*  sh0  = (float* )(ws + alloc(INF_ * 4));
    float*  sc1  = (float* )(ws + alloc(H1_ * 4));
    float*  sh1  = (float* )(ws + alloc(H1_ * 4));
    float*  sc2  = (float* )(ws + alloc(H2_ * 4));
    float*  sh2  = (float* )(ws + alloc(H2_ * 4));

    float* out_ls = (float*)d_out;                         // [16384,1000]
    float* out_h1 = (float*)d_out + (size_t)BB * NC_;      // [16384,2048]
    const float invM = 1.f / (float)BB;

    // 1. BN0 statistics
    colstats_partial<<<dim3(INF_ / 256, 64), 256, 0, stream>>>(x, INF_, BB / 64, psum, psq);
    colstats_final<<<INF_ / 256, 256, 0, stream>>>(psum, psq, 64, INF_, invM, norm0_g, norm0_b, sc0, sh0);

    // 2. fused BN0 + MM/STE block -> v (bf16)
    mm_block<<<(BB * C_) / 256, 256, 0, stream>>>(x, sc0, sh0, S, Hm, T, LUT, bufA);

    // 3. weights -> bf16 (W3 zero-padded to 1024 rows)
    convert_pad<<<((long)H1_ * INF_) / 256, 256, 0, stream>>>(W1, w1b, H1_, INF_, (long)H1_ * INF_);
    convert_pad<<<((long)H2_ * H1_) / 256, 256, 0, stream>>>(W2, w2b, H2_, H1_, (long)H2_ * H1_);
    convert_pad<<<((long)NCP_ * H2_) / 256, 256, 0, stream>>>(W3, w3b, NC_, H2_, (long)NCP_ * H2_);

    // 4. y1 = v @ W1^T + b1
    wmma_gemm_bf16<<<dim3(H1_ / 128, BB / 128), 256, 0, stream>>>(bufA, w1b, bufB, INF_, H1_, b1, H1_);

    // 5. BN1 stats + ReLU; h1 -> d_out (f32) and bf16
    colstats_partial<<<dim3(H1_ / 256, 64), 256, 0, stream>>>(bufB, H1_, BB / 64, psum, psq);
    colstats_final<<<H1_ / 256, 256, 0, stream>>>(psum, psq, 64, H1_, invM, norm1_g, norm1_b, sc1, sh1);
    bnrelu<<<((long)BB * H1_) / 256, 256, 0, stream>>>(bufB, sc1, sh1, H1_, (long)BB * H1_, out_h1, (__bf16*)bufC);

    // 6. y2 = h1 @ W2^T + b2   (reuse bufB)
    wmma_gemm_bf16<<<dim3(H2_ / 128, BB / 128), 256, 0, stream>>>((__bf16*)bufC, w2b, bufB, H1_, H2_, b2, H2_);

    // 7. BN2 stats + ReLU; h2 bf16 -> bufA (reuse)
    colstats_partial<<<dim3(H2_ / 256, 64), 256, 0, stream>>>(bufB, H2_, BB / 64, psum, psq);
    colstats_final<<<H2_ / 256, 256, 0, stream>>>(psum, psq, 64, H2_, invM, norm2_g, norm2_b, sc2, sh2);
    bnrelu<<<((long)BB * H2_) / 256, 256, 0, stream>>>(bufB, sc2, sh2, H2_, (long)BB * H2_, nullptr, bufA);

    // 8. y3 = h2 @ W3^T + b3 (padded N=1024; bias only for n<1000) -> bufC (f32)
    wmma_gemm_bf16<<<dim3(NCP_ / 128, BB / 128), 256, 0, stream>>>(bufA, w3b, (float*)bufC, H2_, NCP_, b3, NC_);

    // 9. log-softmax over first 1000 columns
    logsoftmax_rows<<<BB, 256, 0, stream>>>((float*)bufC, NCP_, out_ls);
}
